// DotProductAttention_63032940036104
// MI455X (gfx1250) — matrix operand order repaired and checked by reading, not compiled, not run on hardware
//
#include <hip/hip_runtime.h>
#include <hip/hip_bf16.h>

// ---------------------------------------------------------------------------
// DotProductAttention on MI455X (gfx1250):
//   S = Q @ K_b        -> bf16 WMMA (16x16x32), f32 accumulate
//   P = softmax_T(S)   -> fp32, wave32-per-row (T=800 = 25*32 exactly)
//   C = P @ K_b^T      -> bf16 WMMA, f32 accumulate
// ~210 GFLOP vs ~0.7 GB traffic => compute bound; bf16 WMMA is the pipe.
// 128x128 block / 32x64 wave tile, double-buffered LDS, 1 barrier per K-step.
// ---------------------------------------------------------------------------

typedef __attribute__((ext_vector_type(16))) __bf16 v16bf;
typedef __attribute__((ext_vector_type(8)))  float  v8f;

#define BM 128
#define BN 128
#define BK 32
#define LDSS 40   // LDS row stride in bf16 elements (80 B, keeps 16B alignment)

union FragU {
    v16bf v;
    uint4 q[2];
};

// A fragment (16x32 bf16, MxK), LDS row-major [row][k], stride LDSS.
// lane 0-15: M=lane,  K 0-7 / 16-23 ; lane 16-31: M=lane-16, K 8-15 / 24-31.
__device__ inline v16bf load_frag_A(const __bf16* base, int lane) {
    const int r  = lane & 15;
    const int hi = lane >> 4;
    const __bf16* p = base + r * LDSS + hi * 8;
    FragU f;
    f.q[0] = *reinterpret_cast<const uint4*>(p);
    f.q[1] = *reinterpret_cast<const uint4*>(p + 16);
    return f.v;
}

// B fragment (32x16 bf16, KxN), LDS stored as [n][k], stride LDSS.
// lane 0-15: N=lane, K 0-7 / 16-23 ; lane 16-31: N=lane-16, K 8-15 / 24-31.
__device__ inline v16bf load_frag_B(const __bf16* base, int lane) {
    const int n  = lane & 15;
    const int hi = lane >> 4;
    const __bf16* p = base + n * LDSS + hi * 8;
    FragU f;
    f.q[0] = *reinterpret_cast<const uint4*>(p);
    f.q[1] = *reinterpret_cast<const uint4*>(p + 16);
    return f.v;
}

__device__ inline void cvt_store16(const float* r, __bf16* dst) {
    __align__(16) __bf16 t[16];
    #pragma unroll
    for (int i = 0; i < 16; ++i) t[i] = (__bf16)r[i];
    *reinterpret_cast<uint4*>(dst)     = reinterpret_cast<const uint4*>(t)[0];
    *reinterpret_cast<uint4*>(dst + 8) = reinterpret_cast<const uint4*>(t)[1];
}

#define WMMA_BF16(A, B, C) \
    __builtin_amdgcn_wmma_f32_16x16x32_bf16(false, (A), false, (B), (short)0, (C), false, false)

// ---------------------------------------------------------------------------
// Generic bf16-WMMA GEMM: C[b] (MxN) = A[b] (MxK, lda=K) * B[b]
//   TRB=true : B memory is KxN row-major (k-major rows)  -> transpose staging
//   TRB=false: B memory is NxK row-major (k-contiguous)  -> direct staging
// ---------------------------------------------------------------------------
template<int M, int N, int K, bool TRB>
__global__ __launch_bounds__(256) void gemm_bf16(
    const float* __restrict__ A, size_t strideA,
    const float* __restrict__ B, size_t strideB,
    float*       __restrict__ C, size_t strideC)
{
    constexpr int NSTEPS = K / BK;
    __shared__ __align__(16) __bf16 sA[2][BM * LDSS];
    __shared__ __align__(16) __bf16 sB[2][BN * LDSS];

    const int b  = blockIdx.z;
    const int n0 = blockIdx.x * BN;
    const int m0 = blockIdx.y * BM;
    const float* Ab = A + (size_t)b * strideA;
    const float* Bb = B + (size_t)b * strideB;
    float*       Cb = C + (size_t)b * strideC;

    const int tid  = threadIdx.x;
    const int lane = tid & 31;
    const int wave = tid >> 5;
    const int wm   = (wave & 3) * 32;   // 4 waves over M
    const int wn   = (wave >> 2) * 64;  // 2 waves over N

    v8f acc[2][4] = {};

    // ---- A staging: 128 rows x 32 k, 16 contiguous floats per thread ----
    const int aRow = tid >> 1;
    const int aKh  = (tid & 1) * 16;
    int aRowG = m0 + aRow; if (aRowG >= M) aRowG = M - 1;
    const float* aSrc = Ab + (size_t)aRowG * K + aKh;
    const int aOff = aRow * LDSS + aKh;

    // ---- B staging maps ----
    int bOff;
    const float* bSrc;
    int bKh;
    if (TRB) {
        const int bN = tid & 127;
        bKh = (tid >> 7) * 16;
        int nG = n0 + bN; if (nG >= N) nG = N - 1;
        bSrc = Bb + nG;                       // element (k,n): bSrc[k * N]
        bOff = bN * LDSS + bKh;
    } else {
        const int bN = tid >> 1;
        bKh = (tid & 1) * 16;
        bSrc = Bb + (size_t)(n0 + bN) * K + bKh;   // 16 contiguous k
        bOff = bN * LDSS + bKh;
    }

    float aReg[16], bReg[16];

    auto loadA = [&](int k0) {
        const float4* p = reinterpret_cast<const float4*>(aSrc + k0);
        #pragma unroll
        for (int q = 0; q < 4; ++q) {
            float4 f = p[q];
            aReg[q*4+0] = f.x; aReg[q*4+1] = f.y; aReg[q*4+2] = f.z; aReg[q*4+3] = f.w;
        }
    };
    auto loadB = [&](int k0) {
        if (TRB) {
            #pragma unroll
            for (int i = 0; i < 16; ++i)
                bReg[i] = bSrc[(size_t)(k0 + bKh + i) * N];
        } else {
            const float4* p = reinterpret_cast<const float4*>(bSrc + k0);
            #pragma unroll
            for (int q = 0; q < 4; ++q) {
                float4 f = p[q];
                bReg[q*4+0] = f.x; bReg[q*4+1] = f.y; bReg[q*4+2] = f.z; bReg[q*4+3] = f.w;
            }
        }
    };

    // ---- prologue: stage step 0 into buffer 0 ----
    loadA(0); loadB(0);
    cvt_store16(aReg, &sA[0][aOff]);
    cvt_store16(bReg, &sB[0][bOff]);
    __syncthreads();

    // ---- main loop: 1 barrier / K-step, global loads shadowed by WMMA ----
    for (int s = 0; s < NSTEPS; ++s) {
        const int cur = s & 1;
        const bool hasNext = (s + 1 < NSTEPS);
        if (hasNext) { loadA((s + 1) * BK); loadB((s + 1) * BK); }

        v16bf a0 = load_frag_A(&sA[cur][(wm +  0) * LDSS], lane);
        v16bf a1 = load_frag_A(&sA[cur][(wm + 16) * LDSS], lane);
        v16bf b0 = load_frag_B(&sB[cur][(wn +  0) * LDSS], lane);
        v16bf b1 = load_frag_B(&sB[cur][(wn + 16) * LDSS], lane);
        v16bf b2 = load_frag_B(&sB[cur][(wn + 32) * LDSS], lane);
        v16bf b3 = load_frag_B(&sB[cur][(wn + 48) * LDSS], lane);

        acc[0][0] = WMMA_BF16(a0, b0, acc[0][0]);
        acc[0][1] = WMMA_BF16(a0, b1, acc[0][1]);
        acc[0][2] = WMMA_BF16(a0, b2, acc[0][2]);
        acc[0][3] = WMMA_BF16(a0, b3, acc[0][3]);
        acc[1][0] = WMMA_BF16(a1, b0, acc[1][0]);
        acc[1][1] = WMMA_BF16(a1, b1, acc[1][1]);
        acc[1][2] = WMMA_BF16(a1, b2, acc[1][2]);
        acc[1][3] = WMMA_BF16(a1, b3, acc[1][3]);

        if (hasNext) {
            cvt_store16(aReg, &sA[cur ^ 1][aOff]);
            cvt_store16(bReg, &sB[cur ^ 1][bOff]);
        }
        __syncthreads();
    }

    // ---- epilogue: fast path for interior blocks (no per-element guards) ----
    const int hi = lane >> 4;
    const int cn = lane & 15;
    if ((m0 + BM <= M) && (n0 + BN <= N)) {
        #pragma unroll
        for (int i = 0; i < 2; ++i) {
            #pragma unroll
            for (int j = 0; j < 4; ++j) {
                float* p = Cb + (size_t)(m0 + wm + i * 16 + hi * 8) * N
                              + (n0 + wn + j * 16 + cn);
                #pragma unroll
                for (int r = 0; r < 8; ++r)
                    p[(size_t)r * N] = acc[i][j][r];   // immediate-offset stores
            }
        }
    } else {
        #pragma unroll
        for (int i = 0; i < 2; ++i) {
            #pragma unroll
            for (int j = 0; j < 4; ++j) {
                const int colG = n0 + wn + j * 16 + cn;
                if (colG >= N) continue;
                const int rowB = m0 + wm + i * 16 + hi * 8;
                #pragma unroll
                for (int r = 0; r < 8; ++r) {
                    if (rowB + r < M)
                        Cb[(size_t)(rowB + r) * N + colG] = acc[i][j][r];
                }
            }
        }
    }
}

// ---------------------------------------------------------------------------
// Softmax over T=800, in place, one wave32 per row (25 elems/lane exactly).
// ---------------------------------------------------------------------------
__global__ __launch_bounds__(256) void softmax_rows(float* __restrict__ W)
{
    const int T = 800;
    const int row  = blockIdx.x * 8 + (threadIdx.x >> 5);
    const int lane = threadIdx.x & 31;
    float* p = W + (size_t)row * T;

    float vals[25];
    float m = -INFINITY;
    #pragma unroll
    for (int i = 0; i < 25; ++i) {
        vals[i] = p[lane + i * 32];
        m = fmaxf(m, vals[i]);
    }
    #pragma unroll
    for (int off = 16; off > 0; off >>= 1)
        m = fmaxf(m, __shfl_xor(m, off, 32));

    float s = 0.f;
    #pragma unroll
    for (int i = 0; i < 25; ++i) {
        vals[i] = __expf(vals[i] - m);
        s += vals[i];
    }
    #pragma unroll
    for (int off = 16; off > 0; off >>= 1)
        s += __shfl_xor(s, off, 32);

    const float inv = 1.0f / s;
    #pragma unroll
    for (int i = 0; i < 25; ++i)
        p[lane + i * 32] = vals[i] * inv;
}

// ---------------------------------------------------------------------------
extern "C" void kernel_launch(void* const* d_in, const int* in_sizes, int n_in,
                              void* d_out, int out_size, void* d_ws, size_t ws_size,
                              hipStream_t stream) {
    (void)in_sizes; (void)n_in; (void)out_size; (void)d_ws; (void)ws_size;
    const int B = 64, V = 1000, D = 1024, T = 800;

    const float* q_embed   = (const float*)d_in[0];   // (V, D)
    const float* conv_feat = (const float*)d_in[1];   // (B, D, T)

    float* context = (float*)d_out;                   // (B, V, D)
    float* weights = context + (size_t)B * V * D;     // (B, V, T)

    dim3 blk(256);

    // GEMM1: S = Q @ K_b.  M=V, N=T, K=D.  B operand is (D,T) k-major -> TRB.
    gemm_bf16<1000, 800, 1024, true>
        <<<dim3((T + BN - 1) / BN, (V + BM - 1) / BM, B), blk, 0, stream>>>(
            q_embed, 0, conv_feat, (size_t)D * T, weights, (size_t)V * T);

    // Softmax over T, in place. 64000 rows, 8 waves per block.
    softmax_rows<<<dim3((B * V) / 8), blk, 0, stream>>>(weights);

    // GEMM2: C = P @ K_b^T.  M=V, N=D, K=T.  B operand rows are k-contiguous.
    gemm_bf16<1000, 1024, 800, false>
        <<<dim3(D / BN, (V + BM - 1) / BM, B), blk, 0, stream>>>(
            weights, (size_t)V * T, conv_feat, (size_t)D * T, context, (size_t)V * D);
}
